// GeometryRep_14456859919018
// MI455X (gfx1250) — matrix operand order repaired and checked by reading, not compiled, not run on hardware
//
#include <hip/hip_runtime.h>
#include <math.h>

typedef _Float16 half_t;
typedef __attribute__((ext_vector_type(16))) _Float16 v16h;
typedef __attribute__((ext_vector_type(8)))  _Float16 v8h;
typedef __attribute__((ext_vector_type(8)))  float    v8f;

#define T_TOK 4096
#define G_GEO 20000

#if __has_builtin(__builtin_amdgcn_global_load_async_to_lds_b128)
#define HAVE_ASYNC_LDS 1
typedef int v4i __attribute__((ext_vector_type(4)));
typedef __attribute__((address_space(1))) v4i* as1_v4i;
typedef __attribute__((address_space(3))) v4i* as3_v4i;
__device__ __forceinline__ void async_copy16(const half_t* g, half_t* l) {
  __builtin_amdgcn_global_load_async_to_lds_b128(
      (as1_v4i)(g), (as3_v4i)(l), 0, 0);
}
#endif

__device__ __forceinline__ void wait_async0() {
#if __has_builtin(__builtin_amdgcn_s_wait_asynccnt)
  __builtin_amdgcn_s_wait_asynccnt(0);
#elif defined(HAVE_ASYNC_LDS)
  asm volatile("s_wait_asynccnt 0x0" ::: "memory");
#endif
}

__device__ __forceinline__ float gelu_erf(float x) {
  return 0.5f * x * (1.0f + erff(x * 0.70710678118654752f));
}

__device__ __forceinline__ v8f wmma_f16(v16h a, v16h b, v8f c) {
  return __builtin_amdgcn_wmma_f32_16x16x32_f16(false, a, false, b,
                                                (short)0, c, false, false);
}

// ---------------------------------------------------------------------------
// Ball query: one thread per token. Keeps K nearest (within radius) sorted
// ascending with jax.top_k tie-breaking (earlier index first). Writes f16
// feature rows [dist, rel(3), rel(3)] * K, zero padded to Kp.
// ---------------------------------------------------------------------------
__global__ void k_ballquery(const float* __restrict__ geo,
                            const float* __restrict__ tok,
                            int G, int T, float r2, int K, int Kp,
                            half_t* __restrict__ feats) {
  int t = blockIdx.x * blockDim.x + threadIdx.x;
  if (t >= T) return;
  float tx = tok[3 * t + 0], ty = tok[3 * t + 1], tz = tok[3 * t + 2];
  float bd[32];
  int bi[32];
  for (int i = 0; i < K; ++i) { bd[i] = INFINITY; bi[i] = 0; }
  for (int g = 0; g < G; ++g) {
    if (g + 128 < G) __builtin_prefetch(geo + 3 * (g + 128), 0, 1);
    float dx = geo[3 * g + 0] - tx;
    float dy = geo[3 * g + 1] - ty;
    float dz = geo[3 * g + 2] - tz;
    float d2 = dx * dx + dy * dy + dz * dz;
    if (d2 <= r2 && d2 < bd[K - 1]) {
      int j = K - 1;
      while (j > 0 && bd[j - 1] > d2) { bd[j] = bd[j - 1]; bi[j] = bi[j - 1]; --j; }
      bd[j] = d2; bi[j] = g;
    }
  }
  size_t base = (size_t)t * Kp;
  for (int kk = 0; kk < K; ++kk) {
    float f0 = 0.f, f1 = 0.f, f2 = 0.f, f3 = 0.f;
    if (bd[kk] <= r2) {
      int g = bi[kk];
      f0 = sqrtf(fmaxf(bd[kk], 1e-12f));
      f1 = geo[3 * g + 0] - tx;
      f2 = geo[3 * g + 1] - ty;
      f3 = geo[3 * g + 2] - tz;
    }
    half_t* fp = feats + base + kk * 7;
    fp[0] = (half_t)f0; fp[1] = (half_t)f1; fp[2] = (half_t)f2; fp[3] = (half_t)f3;
    fp[4] = (half_t)f1; fp[5] = (half_t)f2; fp[6] = (half_t)f3;
  }
  for (int i = K * 7; i < Kp; ++i) feats[base + i] = (half_t)0.0f;
}

// ---------------------------------------------------------------------------
// Transpose + f32->f16 + zero-pad: W is (K x N) row-major f32, out is
// (Np x Kp) row-major f16 with out[n*Kp+k] = W[k*N+n].
// ---------------------------------------------------------------------------
__global__ void k_prep_bt(const float* __restrict__ W, int K, int N,
                          half_t* __restrict__ out, int Kp, int Np) {
  int idx = blockIdx.x * blockDim.x + threadIdx.x;
  int total = Kp * Np;
  if (idx >= total) return;
  int k = idx % Kp;
  int n = idx / Kp;
  float v = (k < K && n < N) ? W[(size_t)k * N + n] : 0.0f;
  out[idx] = (half_t)v;
}

__global__ void k_fill_h(half_t* p, int n) {
  int i = blockIdx.x * blockDim.x + threadIdx.x;
  if (i < n) p[i] = (half_t)0.0f;
}
__global__ void k_fill_f(float* p, int n) {
  int i = blockIdx.x * blockDim.x + threadIdx.x;
  if (i < n) p[i] = 0.0f;
}

// ---------------------------------------------------------------------------
// Small/odd-shape WMMA GEMM (one 16x16 tile per wave, direct global loads).
// A f16 row-major (lda); Bt f16 = B transposed (N x K) row-major (ldb).
// Fragments use contiguous k order (k = 16*half + s) for both A and B.
// ---------------------------------------------------------------------------
__global__ void k_gemm_wmma(const half_t* __restrict__ A, int lda,
                            const half_t* __restrict__ Bt, int ldb,
                            const float* __restrict__ bias,
                            const float* __restrict__ res, int ldres,
                            float* __restrict__ Cf, half_t* __restrict__ Ch,
                            int ldc, int M, int N, int Nreal, int K, int act) {
  int wave = threadIdx.x >> 5;
  int lane = threadIdx.x & 31;
  int tilesM = M >> 4;
  int tilesN = N >> 4;
  int tileIdx = blockIdx.x * (blockDim.x >> 5) + wave;
  if (tileIdx >= tilesM * tilesN) return;  // wave-uniform exit, EXEC stays full
  int tm = tileIdx % tilesM;
  int tn = tileIdx / tilesM;
  int r = lane & 15;
  int hs = lane >> 4;

  v8f acc = {};
  const half_t* arow = A + (size_t)(tm * 16 + r) * lda + 16 * hs;
  const half_t* brow = Bt + (size_t)(tn * 16 + r) * ldb + 16 * hs;

  for (int k0 = 0; k0 < K; k0 += 32) {
    if (k0 + 128 < K) {
      __builtin_prefetch(arow + k0 + 128, 0, 1);
      __builtin_prefetch(brow + k0 + 128, 0, 1);
    }
    v8h alo = *(const v8h*)(arow + k0);
    v8h ahi = *(const v8h*)(arow + k0 + 8);
    v8h blo = *(const v8h*)(brow + k0);
    v8h bhi = *(const v8h*)(brow + k0 + 8);
    v16h a, b;
#pragma unroll
    for (int i = 0; i < 8; ++i) { a[i] = alo[i]; a[8 + i] = ahi[i]; }
#pragma unroll
    for (int i = 0; i < 8; ++i) { b[i] = blo[i]; b[8 + i] = bhi[i]; }
    acc = wmma_f16(a, b, acc);
  }

  int col = tn * 16 + r;
  if (col < Nreal) {
    float bv = bias ? bias[col] : 0.0f;
#pragma unroll
    for (int i = 0; i < 8; ++i) {
      int row = tm * 16 + hs * 8 + i;
      float v = acc[i] + bv;
      if (res) v += res[(size_t)row * ldres + col];
      if (act == 1) v = gelu_erf(v);
      if (Cf) Cf[(size_t)row * ldc + col] = v;
      if (Ch) Ch[(size_t)row * ldc + col] = (half_t)v;
    }
  }
}

// ---------------------------------------------------------------------------
// Big WMMA GEMM: 64x64 block tile, 128 threads = 4 waves, each wave a 32x32
// register tile (2x2 WMMA accumulators). A/Bt chunks (64 rows x 32 halfs)
// are staged into double-buffered LDS via GLOBAL_LOAD_ASYNC_TO_LDS_B128
// (ASYNCcnt + s_wait_asynccnt), overlapping copies with WMMA math.
// LDS rows padded to 40 halfs (80B pitch) to avoid bank conflicts.
// Requires M%64==0, N%64==0, K%32==0.
// ---------------------------------------------------------------------------
#define SRD 40

__global__ void __launch_bounds__(128)
k_gemm_wmma_big(const half_t* __restrict__ A, int lda,
                const half_t* __restrict__ Bt, int ldb,
                const float* __restrict__ bias,
                const float* __restrict__ res, int ldres,
                float* __restrict__ Cf, half_t* __restrict__ Ch,
                int ldc, int N, int Nreal, int K, int act) {
  __shared__ __align__(16) half_t sA[2][64 * SRD];
  __shared__ __align__(16) half_t sB[2][64 * SRD];
  int tid = threadIdx.x;
  int wave = tid >> 5, lane = tid & 31;
  int tilesN = N >> 6;
  int tm64 = blockIdx.x / tilesN;
  int tn64 = blockIdx.x % tilesN;

  // staging: 128 threads x (2 rows of A + 2 rows of B), 16B (8 halfs) each
  int seg = tid & 3;
  int row0 = tid >> 2;  // 0..31
  const half_t* Ag0 = A + (size_t)(tm64 * 64 + row0) * lda + seg * 8;
  const half_t* Ag1 = Ag0 + (size_t)32 * lda;
  const half_t* Bg0 = Bt + (size_t)(tn64 * 64 + row0) * ldb + seg * 8;
  const half_t* Bg1 = Bg0 + (size_t)32 * ldb;
  int la0 = row0 * SRD + seg * 8;
  int la1 = (row0 + 32) * SRD + seg * 8;

  // compute-side fragment bases
  int r = lane & 15, hs = lane >> 4;
  int wm = wave & 1, wn = wave >> 1;
  int ab0 = (wm * 32 + r) * SRD;
  int ab1 = (wm * 32 + 16 + r) * SRD;
  int bb0 = (wn * 32 + r) * SRD;
  int bb1 = (wn * 32 + 16 + r) * SRD;

  v8f acc00 = {}, acc01 = {}, acc10 = {}, acc11 = {};

  auto stage = [&](int buf, int k0) {
#ifdef HAVE_ASYNC_LDS
    async_copy16(Ag0 + k0, &sA[buf][la0]);
    async_copy16(Ag1 + k0, &sA[buf][la1]);
    async_copy16(Bg0 + k0, &sB[buf][la0]);
    async_copy16(Bg1 + k0, &sB[buf][la1]);
#else
    *(v8h*)&sA[buf][la0] = *(const v8h*)(Ag0 + k0);
    *(v8h*)&sA[buf][la1] = *(const v8h*)(Ag1 + k0);
    *(v8h*)&sB[buf][la0] = *(const v8h*)(Bg0 + k0);
    *(v8h*)&sB[buf][la1] = *(const v8h*)(Bg1 + k0);
#endif
  };

  int nk = K >> 5;
  stage(0, 0);
  for (int kc = 0; kc < nk; ++kc) {
    int cur = kc & 1;
    wait_async0();
    __syncthreads();  // buf[cur] fully staged; prev-iter reads complete
    if (kc + 1 < nk) stage(1 - cur, (kc + 1) * 32);  // overlap with math below

    const half_t* pa = sA[cur];
    const half_t* pb = sB[cur];
    v16h af0, af1, bf0, bf1;
    {
      v8h lo = *(const v8h*)&pa[ab0 + 16 * hs];
      v8h hi = *(const v8h*)&pa[ab0 + 16 * hs + 8];
#pragma unroll
      for (int i = 0; i < 8; ++i) { af0[i] = lo[i]; af0[8 + i] = hi[i]; }
      lo = *(const v8h*)&pa[ab1 + 16 * hs];
      hi = *(const v8h*)&pa[ab1 + 16 * hs + 8];
#pragma unroll
      for (int i = 0; i < 8; ++i) { af1[i] = lo[i]; af1[8 + i] = hi[i]; }
    }
    {
      v8h lo = *(const v8h*)&pb[bb0 + 16 * hs];
      v8h hi = *(const v8h*)&pb[bb0 + 16 * hs + 8];
#pragma unroll
      for (int i = 0; i < 8; ++i) { bf0[i] = lo[i]; bf0[8 + i] = hi[i]; }
      lo = *(const v8h*)&pb[bb1 + 16 * hs];
      hi = *(const v8h*)&pb[bb1 + 16 * hs + 8];
#pragma unroll
      for (int i = 0; i < 8; ++i) { bf1[i] = lo[i]; bf1[8 + i] = hi[i]; }
    }
    acc00 = wmma_f16(af0, bf0, acc00);
    acc01 = wmma_f16(af0, bf1, acc01);
    acc10 = wmma_f16(af1, bf0, acc10);
    acc11 = wmma_f16(af1, bf1, acc11);
    __syncthreads();  // reads of buf[cur] done before it is restaged
  }

  // epilogue: 2x2 tiles of 16x16
#pragma unroll
  for (int ti = 0; ti < 2; ++ti) {
#pragma unroll
    for (int tj = 0; tj < 2; ++tj) {
      v8f acc = (ti == 0) ? (tj == 0 ? acc00 : acc01)
                          : (tj == 0 ? acc10 : acc11);
      int col = tn64 * 64 + wn * 32 + tj * 16 + r;
      if (col < Nreal) {
        float bv = bias ? bias[col] : 0.0f;
#pragma unroll
        for (int i = 0; i < 8; ++i) {
          int row = tm64 * 64 + wm * 32 + ti * 16 + hs * 8 + i;
          float v = acc[i] + bv;
          if (res) v += res[(size_t)row * ldres + col];
          if (act == 1) v = gelu_erf(v);
          if (Cf) Cf[(size_t)row * ldc + col] = v;
          if (Ch) Ch[(size_t)row * ldc + col] = (half_t)v;
        }
      }
    }
  }
}

// ---------------------------------------------------------------------------
// LayerNorm over 256 cols: one wave per token, wave32 shfl reductions.
// ---------------------------------------------------------------------------
__global__ void k_layernorm(const float* __restrict__ x,
                            const float* __restrict__ g,
                            const float* __restrict__ b,
                            half_t* __restrict__ out16, int ncols) {
  int wave = threadIdx.x >> 5;
  int lane = threadIdx.x & 31;
  int t = blockIdx.x * (blockDim.x >> 5) + wave;
  const float* row = x + (size_t)t * ncols;
  float s = 0.f, s2 = 0.f;
  for (int i = lane; i < ncols; i += 32) {
    float v = row[i];
    s += v; s2 += v * v;
  }
  for (int m = 16; m; m >>= 1) {
    s += __shfl_xor(s, m, 32);
    s2 += __shfl_xor(s2, m, 32);
  }
  float mean = s / ncols;
  float var = fmaxf(s2 / ncols - mean * mean, 0.0f);
  float inv = rsqrtf(var + 1e-5f);
  for (int i = lane; i < ncols; i += 32) {
    float v = (row[i] - mean) * inv * g[i] + b[i];
    out16[(size_t)t * ncols + i] = (half_t)v;
  }
}

// ---------------------------------------------------------------------------
// Slice softmax: block = one token (128 threads = 4 waves = 4 heads),
// lane s computes logit for slice s; softmax across the wave.
// ---------------------------------------------------------------------------
__global__ void k_slice_softmax(const float* __restrict__ xm,
                                const float* __restrict__ wslice,
                                const float* __restrict__ bslice,
                                const float* __restrict__ temp,
                                half_t* __restrict__ swT,
                                half_t* __restrict__ swN,
                                float* __restrict__ normf, int T) {
  int t = blockIdx.x;
  int h = threadIdx.x >> 5;
  int s = threadIdx.x & 31;
  const float* xr = xm + (size_t)t * 256 + h * 64;
  float dot = 0.f;
#pragma unroll 8
  for (int d = 0; d < 64; ++d) dot += xr[d] * wslice[d * 32 + s];
  float tv = fminf(fmaxf(temp[h], 0.1f), 5.0f);
  float l = (dot + bslice[s]) / tv;
  float m = l;
  for (int k = 16; k; k >>= 1) m = fmaxf(m, __shfl_xor(m, k, 32));
  float e = expf(l - m);
  float ssum = e;
  for (int k = 16; k; k >>= 1) ssum += __shfl_xor(ssum, k, 32);
  float sw = e / ssum;
  swT[((size_t)h * 32 + s) * T + t] = (half_t)sw;
  swN[((size_t)h * T + t) * 32 + s] = (half_t)sw;
  atomicAdd(normf + h * 32 + s, sw);
}

// ---------------------------------------------------------------------------
// Tiny per-head slice attention (normalize st, q/k/v, 32x32 softmax attn,
// ot = attn@v), writes otT (d-major f16) for the deslice WMMA GEMM.
// ---------------------------------------------------------------------------
__global__ void k_slice_attn(const float* __restrict__ straw,
                             const float* __restrict__ normf,
                             const float* __restrict__ wq, const float* __restrict__ bq,
                             const float* __restrict__ wk, const float* __restrict__ bk,
                             const float* __restrict__ wv, const float* __restrict__ bv,
                             half_t* __restrict__ otT) {
  int h = blockIdx.x;
  __shared__ float st[32 * 64], qs[32 * 64], ks[32 * 64], vs[32 * 64], att[32 * 32];
  const float* stg = straw + h * 2048;
  const float* nr = normf + h * 32;
  int tid = threadIdx.x;
  for (int i = tid; i < 2048; i += 256) {
    int s = i >> 6;
    st[i] = stg[i] / (nr[s] + 1e-5f);
  }
  __syncthreads();
  for (int i = tid; i < 2048; i += 256) {
    int s = i >> 6, d = i & 63;
    float aq = bq[d], ak = bk[d], av = bv[d];
    for (int e = 0; e < 64; ++e) {
      float x = st[s * 64 + e];
      aq += x * wq[e * 64 + d];
      ak += x * wk[e * 64 + d];
      av += x * wv[e * 64 + d];
    }
    qs[i] = aq; ks[i] = ak; vs[i] = av;
  }
  __syncthreads();
  for (int i = tid; i < 1024; i += 256) {
    int s = i >> 5, u = i & 31;
    float a = 0.f;
    for (int e = 0; e < 64; ++e) a += qs[s * 64 + e] * ks[u * 64 + e];
    att[i] = a * 0.125f;  // 1/sqrt(64)
  }
  __syncthreads();
  if (tid < 32) {
    int s = tid;
    float m = -INFINITY;
    for (int u = 0; u < 32; ++u) m = fmaxf(m, att[s * 32 + u]);
    float sum = 0.f;
    for (int u = 0; u < 32; ++u) { float e = expf(att[s * 32 + u] - m); att[s * 32 + u] = e; sum += e; }
    float inv = 1.0f / sum;
    for (int u = 0; u < 32; ++u) att[s * 32 + u] *= inv;
  }
  __syncthreads();
  for (int i = tid; i < 2048; i += 256) {
    int s = i >> 6, d = i & 63;
    float a = 0.f;
    for (int u = 0; u < 32; ++u) a += att[s * 32 + u] * vs[u * 64 + d];
    otT[(size_t)h * 64 * 32 + d * 32 + s] = (half_t)a;
  }
}

// ---------------------------------------------------------------------------
// Host orchestration
// ---------------------------------------------------------------------------
extern "C" void kernel_launch(void* const* d_in, const int* in_sizes, int n_in,
                              void* d_out, int out_size, void* d_ws, size_t ws_size,
                              hipStream_t stream) {
  (void)in_sizes; (void)n_in; (void)out_size; (void)ws_size;
  const int T = T_TOK;
  auto F = [&](int i) { return (const float*)d_in[i]; };

  char* wsp = (char*)d_ws;
  size_t off = 0;
  auto alloc = [&](size_t bytes) -> void* {
    off = (off + 255) & ~(size_t)255;
    void* p = wsp + off;
    off += bytes;
    return p;
  };

  const int Kn[3] = {8, 16, 32};
  const int Kpad[3] = {64, 128, 224};
  const float rr[3] = {0.05f, 0.1f, 0.25f};

  half_t* w1t[3]; half_t* w2t[3]; half_t* w3t[3];
  for (int j = 0; j < 3; ++j) {
    w1t[j] = (half_t*)alloc((size_t)256 * Kpad[j] * 2);
    w2t[j] = (half_t*)alloc((size_t)128 * 256 * 2);
    w3t[j] = (half_t*)alloc((size_t)96 * 128 * 2);
  }
  half_t* mergewt = (half_t*)alloc((size_t)256 * 256 * 2);
  half_t* wfxt[2]; half_t* wxt[2]; half_t* wot[2]; half_t* wm1t[2]; half_t* wm2t[2];
  for (int b = 0; b < 2; ++b) {
    wfxt[b] = (half_t*)alloc((size_t)256 * 256 * 2);
    wxt[b]  = (half_t*)alloc((size_t)256 * 256 * 2);
    wot[b]  = (half_t*)alloc((size_t)256 * 256 * 2);
    wm1t[b] = (half_t*)alloc((size_t)1024 * 256 * 2);
    wm2t[b] = (half_t*)alloc((size_t)256 * 1024 * 2);
  }
  half_t* wlastt = (half_t*)alloc((size_t)256 * 256 * 2);

  half_t* feats[3];
  for (int j = 0; j < 3; ++j) feats[j] = (half_t*)alloc((size_t)T * Kpad[j] * 2);
  half_t* h1_16 = (half_t*)alloc((size_t)T * 256 * 2);
  half_t* h2_16 = (half_t*)alloc((size_t)T * 128 * 2);
  half_t* concat16 = (half_t*)alloc((size_t)T * 256 * 2);
  float*  xbuf  = (float*)alloc((size_t)T * 256 * 4);
  half_t* hln16 = (half_t*)alloc((size_t)T * 256 * 2);
  float*  fxf   = (float*)alloc((size_t)T * 256 * 4);
  float*  xmf   = (float*)alloc((size_t)T * 256 * 4);
  half_t* fxT16 = (half_t*)alloc((size_t)256 * T * 2);
  half_t* swT16 = (half_t*)alloc((size_t)4 * 32 * T * 2);
  half_t* swN16 = (half_t*)alloc((size_t)4 * T * 32 * 2);
  float*  normf = (float*)alloc(128 * 4);
  float*  straw = (float*)alloc((size_t)4 * 32 * 64 * 4);
  half_t* otT16 = (half_t*)alloc((size_t)4 * 64 * 32 * 2);
  half_t* desl16 = (half_t*)alloc((size_t)T * 256 * 2);
  half_t* mh16  = (half_t*)alloc((size_t)T * 1024 * 2);

  auto prep = [&](const float* W, int K, int N, half_t* out, int Kp, int Np) {
    int total = Kp * Np;
    k_prep_bt<<<(total + 255) / 256, 256, 0, stream>>>(W, K, N, out, Kp, Np);
  };
  auto gemm = [&](const half_t* A, int lda, const half_t* Bt, int ldb,
                  const float* bias, const float* res, int ldres,
                  float* Cf, half_t* Ch, int ldc, int M, int N, int Nreal,
                  int K, int act) {
    if ((M & 63) == 0 && (N & 63) == 0) {
      int grid = (M >> 6) * (N >> 6);
      k_gemm_wmma_big<<<grid, 128, 0, stream>>>(A, lda, Bt, ldb, bias, res,
                                                ldres, Cf, Ch, ldc, N, Nreal,
                                                K, act);
    } else {
      int tiles = (M >> 4) * (N >> 4);
      int grid = (tiles + 3) / 4;
      k_gemm_wmma<<<grid, 128, 0, stream>>>(A, lda, Bt, ldb, bias, res, ldres,
                                            Cf, Ch, ldc, M, N, Nreal, K, act);
    }
  };
  auto lnorm = [&](const float* x, const float* g, const float* b, half_t* o) {
    k_layernorm<<<T / 8, 256, 0, stream>>>(x, g, b, o, 256);
  };

  const float* geo = F(0);
  const float* tok = F(1);

  // ---- ball query features (3 scales) ----
  for (int j = 0; j < 3; ++j) {
    k_ballquery<<<(T + 255) / 256, 256, 0, stream>>>(
        geo, tok, G_GEO, T, rr[j] * rr[j], Kn[j], Kpad[j], feats[j]);
  }

  // ---- prep weights ----
  for (int j = 0; j < 3; ++j) {
    int base = 2 + 6 * j;
    prep(F(base + 0), Kn[j] * 7, 256, w1t[j], Kpad[j], 256);
    prep(F(base + 2), 256, 128, w2t[j], 256, 128);
    prep(F(base + 4), 128, 85, w3t[j], 128, 96);
  }
  prep(F(20), 255, 256, mergewt, 256, 256);
  for (int b = 0; b < 2; ++b) {
    int base = 22 + 23 * b;
    prep(F(base + 2), 256, 256, wfxt[b], 256, 256);
    prep(F(base + 4), 256, 256, wxt[b], 256, 256);
    prep(F(base + 15), 256, 256, wot[b], 256, 256);
    prep(F(base + 19), 256, 1024, wm1t[b], 256, 1024);
    prep(F(base + 21), 1024, 256, wm2t[b], 1024, 256);
  }
  prep(F(70), 256, 256, wlastt, 256, 256);

  // ---- per-scale MLPs -> concat buffer ----
  k_fill_h<<<(T * 256 + 255) / 256, 256, 0, stream>>>(concat16, T * 256);
  const int colofs[3] = {0, 85, 170};
  for (int j = 0; j < 3; ++j) {
    int base = 2 + 6 * j;
    gemm(feats[j], Kpad[j], w1t[j], Kpad[j], F(base + 1), nullptr, 0,
         nullptr, h1_16, 256, T, 256, 256, Kpad[j], 1);
    gemm(h1_16, 256, w2t[j], 256, F(base + 3), nullptr, 0,
         nullptr, h2_16, 128, T, 128, 128, 256, 1);
    gemm(h2_16, 128, w3t[j], 128, F(base + 5), nullptr, 0,
         nullptr, concat16 + colofs[j], 256, T, 96, 85, 128, 0);
  }

  // ---- merge ----
  gemm(concat16, 256, mergewt, 256, F(21), nullptr, 0,
       xbuf, nullptr, 256, T, 256, 256, 256, 0);

  // ---- transolver blocks ----
  for (int b = 0; b < 2; ++b) {
    int base = 22 + 23 * b;
    lnorm(xbuf, F(base + 0), F(base + 1), hln16);
    gemm(hln16, 256, wfxt[b], 256, F(base + 3), nullptr, 0,
         fxf, nullptr, 256, T, 256, 256, 256, 0);
    gemm(hln16, 256, wxt[b], 256, F(base + 5), nullptr, 0,
         xmf, nullptr, 256, T, 256, 256, 256, 0);
    prep(fxf, T, 256, fxT16, T, 256);
    k_fill_f<<<1, 128, 0, stream>>>(normf, 128);
    k_slice_softmax<<<T, 128, 0, stream>>>(xmf, F(base + 6), F(base + 7),
                                           F(base + 8), swT16, swN16, normf, T);
    for (int h = 0; h < 4; ++h)
      gemm(swT16 + (size_t)h * 32 * T, T, fxT16 + (size_t)h * 64 * T, T,
           nullptr, nullptr, 0, straw + h * 2048, nullptr, 64,
           32, 64, 64, T, 0);
    k_slice_attn<<<4, 256, 0, stream>>>(straw, normf,
                                        F(base + 9), F(base + 10),
                                        F(base + 11), F(base + 12),
                                        F(base + 13), F(base + 14), otT16);
    for (int h = 0; h < 4; ++h)
      gemm(swN16 + (size_t)h * T * 32, 32, otT16 + (size_t)h * 64 * 32, 32,
           nullptr, nullptr, 0, nullptr, desl16 + h * 64, 256,
           T, 64, 64, 32, 0);
    gemm(desl16, 256, wot[b], 256, F(base + 16), xbuf, 256,
         xbuf, nullptr, 256, T, 256, 256, 256, 0);
    lnorm(xbuf, F(base + 17), F(base + 18), hln16);
    gemm(hln16, 256, wm1t[b], 256, F(base + 20), nullptr, 0,
         nullptr, mh16, 1024, T, 1024, 1024, 256, 1);
    gemm(mh16, 1024, wm2t[b], 1024, F(base + 22), xbuf, 256,
         xbuf, nullptr, 256, T, 256, 256, 1024, 0);
  }

  // ---- final LN + projection ----
  lnorm(xbuf, F(68), F(69), hln16);
  gemm(hln16, 256, wlastt, 256, F(71), nullptr, 0,
       (float*)d_out, nullptr, 256, T, 256, 256, 256, 0);
}
